// PagedLLaMA_62414464745818
// MI455X (gfx1250) — matrix operand order repaired and checked by reading, not compiled, not run on hardware
//
#include <hip/hip_runtime.h>

// ---------------------------------------------------------------- constants
constexpr int S_LEN = 2048;
constexpr int DMODEL = 4096;
constexpr int NH = 32;
constexpr int NKVH = 8;
constexpr int DHEAD = 128;
constexpr int HID = 11008;
constexpr float EPS = 1e-5f;
constexpr float ROPE_BASE = 10000.0f;

// ---------------------------------------------------------------- types
typedef __attribute__((ext_vector_type(16))) __bf16 v16bf;
typedef __attribute__((ext_vector_type(2)))  __bf16 v2bf;
typedef __attribute__((ext_vector_type(8)))  float  v8f;
typedef __attribute__((ext_vector_type(8)))  unsigned short us8;
typedef __attribute__((ext_vector_type(4)))  unsigned int   ui4;

union FragB16 {
    v16bf v;
    us8   h[2];
    unsigned short u[16];
};

// scalar f32 -> bf16 (hardware convert, RNE)
__device__ __forceinline__ unsigned short f2bf(float f) {
    return __builtin_bit_cast(unsigned short, (__bf16)f);
}

// packed 2x f32 -> 1 dword of bf16 (V_CVT_PK_BF16_F32 when available)
__device__ __forceinline__ unsigned int f2bf2(float lo, float hi) {
#if __has_builtin(__builtin_amdgcn_cvt_pk_bf16_f32)
    return __builtin_bit_cast(unsigned int, __builtin_amdgcn_cvt_pk_bf16_f32(lo, hi));
#else
    v2bf v = { (__bf16)lo, (__bf16)hi };
    return __builtin_bit_cast(unsigned int, v);
#endif
}

__device__ __forceinline__ v8f wmma_bf16(const FragB16& a, const FragB16& b, v8f c) {
    // D = A(16x32 bf16) * B(32x16 bf16) + C(16x16 f32)
    return __builtin_amdgcn_wmma_f32_16x16x32_bf16(
        /*neg_a=*/false, a.v, /*neg_b=*/false, b.v,
        /*c_mod=*/(short)0, c, /*reuse_a=*/false, /*reuse_b=*/false);
}

// ---------------------------------------------------------------- rmsnorm -> bf16
__global__ __launch_bounds__(256)
void rmsnorm_bf16_kernel(const float* __restrict__ x, const float* __restrict__ w,
                         unsigned short* __restrict__ out, int ncols) {
    __shared__ float red[256];
    const int row = blockIdx.x;
    const float* xr = x + (size_t)row * ncols;
    float s = 0.f;
    for (int c = threadIdx.x; c < ncols; c += 256) { float v = xr[c]; s += v * v; }
    red[threadIdx.x] = s;
    __syncthreads();
    for (int st = 128; st > 0; st >>= 1) {
        if ((int)threadIdx.x < st) red[threadIdx.x] += red[threadIdx.x + st];
        __syncthreads();
    }
    const float rs = rsqrtf(red[0] / (float)ncols + EPS);
    unsigned short* orow = out + (size_t)row * ncols;
    for (int c = threadIdx.x; c < ncols; c += 256) orow[c] = f2bf(xr[c] * rs * w[c]);
}

// ---------------------------------------------------------------- tiled WMMA GEMM
// C[M,N] (f32) = A[M,K] (bf16, row-major) * Bw[K,N] (f32, row-major, converted on the fly)
//              (+ Cin[M,N] if Cin != nullptr)
// Double-buffered LDS; B staged in fragment-major (column) layout so B-fragment
// reads are 2x ds_load_b128 per fragment, same as A.
#define BM 128
#define BN 128
#define BK 32
#define LSTR 40    // ushort stride, 80B: 16B aligned

__global__ __launch_bounds__(256)
void gemm_bf16_wmma_kernel(const unsigned short* __restrict__ A,
                           const float* __restrict__ Bw,
                           const float* __restrict__ Cin,
                           float* __restrict__ Cout,
                           int M, int N, int K) {
    __shared__ unsigned short As[2][BM][LSTR];   // [row][k]
    __shared__ unsigned short Bs[2][BN][LSTR];   // [col][k]  (column-major tile)

    const int tid  = threadIdx.x;
    const int lane = tid & 31;
    const int wave = tid >> 5;            // 0..7
    const int wm   = wave >> 2;           // 0..1  (64-row strip)
    const int wn   = wave & 3;            // 0..3  (32-col strip)
    const int l16  = lane & 15;
    const int lhi  = lane >> 4;

    const int row0 = blockIdx.y * BM;
    const int col0 = blockIdx.x * BN;

    const v8f vzero = {0.f, 0.f, 0.f, 0.f, 0.f, 0.f, 0.f, 0.f};
    v8f acc[4][2];
#pragma unroll
    for (int mi = 0; mi < 4; ++mi)
#pragma unroll
        for (int ni = 0; ni < 2; ++ni) acc[mi][ni] = vzero;

    // A staging: thread -> (row = tid/2, k-half = tid&1), 16 contiguous ushorts
    const int ar = tid >> 1;
    const int ac = (tid & 1) * 16;
    const unsigned short* aptr = A + (size_t)(row0 + ar) * K + ac;

    // B staging (column gather): thread -> (col = tid&127, k-half = tid>>7);
    // 16 global_load_b32, lane-coalesced along N.
    const int bn_ = tid & 127;
    const int bkh = tid >> 7;
    const float* bptr = Bw + (size_t)(bkh * 16) * N + col0 + bn_;

    const int ntiles = K / BK;

    // ---- preload tile 0
    us8 a0v = *(const us8*)(aptr);
    us8 a1v = *(const us8*)(aptr + 8);
    float bv[16];
#pragma unroll
    for (int j = 0; j < 16; ++j) bv[j] = bptr[(size_t)j * N];

    {   // store tile 0 into buffer 0 (packed hardware bf16 converts)
        *(us8*)&As[0][ar][ac]     = a0v;
        *(us8*)&As[0][ar][ac + 8] = a1v;
        ui4 q0, q1;
#pragma unroll
        for (int j = 0; j < 4; ++j) {
            q0[j] = f2bf2(bv[2 * j],     bv[2 * j + 1]);
            q1[j] = f2bf2(bv[2 * j + 8], bv[2 * j + 9]);
        }
        *(ui4*)&Bs[0][bn_][bkh * 16]     = q0;
        *(ui4*)&Bs[0][bn_][bkh * 16 + 8] = q1;
    }

    int buf = 0;
    for (int kt = 0; kt < ntiles; ++kt) {
        __syncthreads();
        const bool more = (kt + 1) < ntiles;

        // ---- prefetch next tile into registers (overlaps with WMMAs below)
        us8 a0n, a1n;
        float bn_v[16];
        if (more) {
            aptr += BK;
            bptr += (size_t)BK * N;
            a0n = *(const us8*)(aptr);
            a1n = *(const us8*)(aptr + 8);
#pragma unroll
            for (int j = 0; j < 16; ++j) bn_v[j] = bptr[(size_t)j * N];
        }

        // ---- fragments from current buffer
        FragB16 af[4], bfg[2];
#pragma unroll
        for (int mi = 0; mi < 4; ++mi) {
            const int r   = wm * 64 + mi * 16 + l16;   // M = lane&15
            const int kof = lhi * 8;                   // K block by lane[4]
            af[mi].h[0] = *(const us8*)&As[buf][r][kof];
            af[mi].h[1] = *(const us8*)&As[buf][r][16 + kof];
        }
#pragma unroll
        for (int ni = 0; ni < 2; ++ni) {
            const int c = wn * 32 + ni * 16 + l16;     // N = lane&15
            bfg[ni].h[0] = *(const us8*)&Bs[buf][c][lhi * 8];
            bfg[ni].h[1] = *(const us8*)&Bs[buf][c][16 + lhi * 8];
        }
#pragma unroll
        for (int mi = 0; mi < 4; ++mi)
#pragma unroll
            for (int ni = 0; ni < 2; ++ni)
                acc[mi][ni] = wmma_bf16(af[mi], bfg[ni], acc[mi][ni]);

        // ---- stash prefetched tile into the other buffer
        if (more) {
            const int nb = buf ^ 1;
            *(us8*)&As[nb][ar][ac]     = a0n;
            *(us8*)&As[nb][ar][ac + 8] = a1n;
            ui4 q0, q1;
#pragma unroll
            for (int j = 0; j < 4; ++j) {
                q0[j] = f2bf2(bn_v[2 * j],     bn_v[2 * j + 1]);
                q1[j] = f2bf2(bn_v[2 * j + 8], bn_v[2 * j + 9]);
            }
            *(ui4*)&Bs[nb][bn_][bkh * 16]     = q0;
            *(ui4*)&Bs[nb][bn_][bkh * 16 + 8] = q1;
        }
        buf ^= 1;
    }

    // ---- epilogue: C/D layout -> lane holds col = l16, rows r + 8*lhi
#pragma unroll
    for (int mi = 0; mi < 4; ++mi)
#pragma unroll
        for (int ni = 0; ni < 2; ++ni) {
            const int cb = col0 + wn * 32 + ni * 16 + l16;
#pragma unroll
            for (int r = 0; r < 8; ++r) {
                const int rr = row0 + wm * 64 + mi * 16 + r + 8 * lhi;
                const size_t off = (size_t)rr * N + cb;
                float v = acc[mi][ni][r];
                if (Cin) v += Cin[off];
                Cout[off] = v;
            }
        }
}

// ---------------------------------------------------------------- RoPE + head transpose -> bf16 [h][s][dh]
__global__ __launch_bounds__(256)
void rope_transpose_kernel(const float* __restrict__ src, unsigned short* __restrict__ dst,
                           const int* __restrict__ pos, int nh) {
    const int half = DHEAD / 2;
    const int total = S_LEN * nh * half;
    const int idx = blockIdx.x * 256 + threadIdx.x;
    if (idx >= total) return;
    const int i = idx % half;
    const int t = idx / half;
    const int h = t % nh;
    const int s = t / nh;
    const float p = (float)pos[s];
    const float freq = __powf(ROPE_BASE, -((float)(2 * i) / (float)DHEAD));
    float sn, cs;
    __sincosf(p * freq, &sn, &cs);
    const float* sp = src + ((size_t)s * nh + h) * DHEAD;
    const float x1 = sp[2 * i], x2 = sp[2 * i + 1];
    unsigned int* dp = (unsigned int*)(dst + ((size_t)h * S_LEN + s) * DHEAD);
    dp[i] = f2bf2(x1 * cs - x2 * sn, x1 * sn + x2 * cs);
}

// V: [s][h][d] f32  ->  [h][d][s] bf16 (so P*V B-fragments are contiguous along keys)
__global__ __launch_bounds__(256)
void v_transpose_kernel(const float* __restrict__ src, unsigned short* __restrict__ dst) {
    const int total = S_LEN * NKVH * DHEAD;
    const int idx = blockIdx.x * 256 + threadIdx.x;
    if (idx >= total) return;
    const int d = idx % DHEAD;
    const int t = idx / DHEAD;
    const int h = t % NKVH;
    const int s = t / NKVH;
    dst[((size_t)h * DHEAD + d) * S_LEN + s] = f2bf(src[idx]);
}

// ---------------------------------------------------------------- flash attention (1 wave per (head, 16-query tile))
__global__ __launch_bounds__(32)
void flash_attn_kernel(const unsigned short* __restrict__ Q,   // [H][S][DH] bf16
                       const unsigned short* __restrict__ Kt,  // [KVH][S][DH] bf16
                       const unsigned short* __restrict__ Vt,  // [KVH][DH][S] bf16
                       unsigned short* __restrict__ O) {       // [S][H*DH] bf16
    __shared__ float sS[16][36];
    __shared__ unsigned short sP[16][40];
    __shared__ float sM[16], sL[16], sA[16];

    const int lane = threadIdx.x;
    const int l16  = lane & 15;
    const int lhi  = lane >> 4;
    const int q0   = blockIdx.x * 16;
    const int h    = blockIdx.y;
    const int kvh  = h >> 2;                      // H / KVH == 4
    const float scale = 0.08838834764831845f;     // 1/sqrt(128)

    // Q fragments for the whole tile (4 x K=32 chunks over DH=128)
    FragB16 qf[4];
    {
        const unsigned short* qrow = Q + ((size_t)h * S_LEN + (q0 + l16)) * DHEAD;
#pragma unroll
        for (int c = 0; c < 4; ++c) {
            const int base = c * 32 + lhi * 8;
            qf[c].h[0] = *(const us8*)(qrow + base);
            qf[c].h[1] = *(const us8*)(qrow + base + 16);
        }
    }

    const v8f vzero = {0.f, 0.f, 0.f, 0.f, 0.f, 0.f, 0.f, 0.f};
    v8f o[8];
#pragma unroll
    for (int n = 0; n < 8; ++n) o[n] = vzero;
    if (lane < 16) { sM[lane] = -1e30f; sL[lane] = 0.f; }
    __syncthreads();

    const int kbmax = (q0 + 15) / 32;
    for (int kb = 0; kb <= kbmax; ++kb) {
        const int kbase = kb * 32;

        // ---- S = Q * K^T  (two 16-key sub-tiles)
        v8f sacc[2] = {vzero, vzero};
#pragma unroll
        for (int sn = 0; sn < 2; ++sn) {
            const unsigned short* krow =
                Kt + ((size_t)kvh * S_LEN + (kbase + sn * 16 + l16)) * DHEAD;
#pragma unroll
            for (int c = 0; c < 4; ++c) {
                FragB16 kf;
                const unsigned short* p = krow + c * 32 + lhi * 8;
                kf.h[0] = *(const us8*)(p);
                kf.h[1] = *(const us8*)(p + 16);
                sacc[sn] = wmma_bf16(qf[c], kf, sacc[sn]);
            }
        }

        // ---- masked, scaled scores to LDS
#pragma unroll
        for (int sn = 0; sn < 2; ++sn) {
            const int key = kbase + sn * 16 + l16;
#pragma unroll
            for (int r = 0; r < 8; ++r) {
                const int qq = q0 + r + 8 * lhi;
                sS[r + 8 * lhi][sn * 16 + l16] =
                    (key <= qq) ? sacc[sn][r] * scale : -1e30f;
            }
        }
        __syncthreads();

        // ---- online softmax (one lane per query row)
        if (lane < 16) {
            float mOld = sM[lane], mNew = mOld;
#pragma unroll
            for (int c = 0; c < 32; ++c) mNew = fmaxf(mNew, sS[lane][c]);
            const float alpha = __expf(mOld - mNew);
            float ps = 0.f;
            unsigned int* prow = (unsigned int*)&sP[lane][0];
#pragma unroll
            for (int c = 0; c < 16; ++c) {
                const float p0 = __expf(sS[lane][2 * c]     - mNew);
                const float p1 = __expf(sS[lane][2 * c + 1] - mNew);
                prow[c] = f2bf2(p0, p1);
                ps += p0 + p1;
            }
            sL[lane] = sL[lane] * alpha + ps;
            sM[lane] = mNew;
            sA[lane] = alpha;
        }
        __syncthreads();

        // ---- rescale O accumulators
#pragma unroll
        for (int r = 0; r < 8; ++r) {
            const float a = sA[r + 8 * lhi];
#pragma unroll
            for (int n = 0; n < 8; ++n) o[n][r] *= a;
        }

        // ---- O += P * V
        FragB16 pf;
        {
            const int kof = lhi * 8;
            pf.h[0] = *(const us8*)&sP[l16][kof];
            pf.h[1] = *(const us8*)&sP[l16][16 + kof];
        }
#pragma unroll
        for (int n = 0; n < 8; ++n) {
            FragB16 vf;
            const unsigned short* p =
                Vt + ((size_t)kvh * DHEAD + n * 16 + l16) * S_LEN + kbase + lhi * 8;
            vf.h[0] = *(const us8*)(p);
            vf.h[1] = *(const us8*)(p + 16);
            o[n] = wmma_bf16(pf, vf, o[n]);
        }
        __syncthreads();
    }

    // ---- normalize and write attn output [s][h*DH + d]
#pragma unroll
    for (int r = 0; r < 8; ++r) {
        const int row = r + 8 * lhi;
        const float inv = 1.f / sL[row];
        unsigned short* op = O + (size_t)(q0 + row) * (NH * DHEAD) + (size_t)h * DHEAD;
#pragma unroll
        for (int n = 0; n < 8; ++n) op[n * 16 + l16] = f2bf(o[n][r] * inv);
    }
}

// ---------------------------------------------------------------- silu(g)*u -> bf16 (2 elems/thread, packed convert)
__global__ __launch_bounds__(256)
void silu_mul_kernel(const float* __restrict__ g, const float* __restrict__ u,
                     unsigned int* __restrict__ out, long long npairs) {
    const long long i = (long long)blockIdx.x * 256 + threadIdx.x;
    if (i >= npairs) return;
    const float x0 = g[2 * i], x1 = g[2 * i + 1];
    const float s0 = x0 / (1.f + __expf(-x0));
    const float s1 = x1 / (1.f + __expf(-x1));
    out[i] = f2bf2(s0 * u[2 * i], s1 * u[2 * i + 1]);
}

// ---------------------------------------------------------------- host launch
extern "C" void kernel_launch(void* const* d_in, const int* in_sizes, int n_in,
                              void* d_out, int out_size, void* d_ws, size_t ws_size,
                              hipStream_t stream) {
    (void)in_sizes; (void)n_in; (void)out_size; (void)ws_size;

    const float* x      = (const float*)d_in[0];
    const int*   pos    = (const int*)  d_in[1];
    const float* ln_w   = (const float*)d_in[2];
    const float* ffln_w = (const float*)d_in[3];
    const float* wq     = (const float*)d_in[4];
    const float* wk     = (const float*)d_in[5];
    const float* wv     = (const float*)d_in[6];
    const float* wo     = (const float*)d_in[7];
    const float* wg     = (const float*)d_in[8];
    const float* w1     = (const float*)d_in[9];
    const float* w2     = (const float*)d_in[10];
    float* out = (float*)d_out;

    char* ws = (char*)d_ws;
    size_t off = 0;
    auto alloc = [&](size_t bytes) -> void* {
        void* p = ws + off;
        off += (bytes + 255) & ~(size_t)255;
        return p;
    };

    unsigned short* a_b   = (unsigned short*)alloc((size_t)S_LEN * DMODEL * 2);
    float*          q_f   = (float*)alloc((size_t)S_LEN * NH   * DHEAD * 4);
    float*          k_f   = (float*)alloc((size_t)S_LEN * NKVH * DHEAD * 4);
    float*          v_f   = (float*)alloc((size_t)S_LEN * NKVH * DHEAD * 4);
    unsigned short* qh_b  = (unsigned short*)alloc((size_t)S_LEN * NH   * DHEAD * 2);
    unsigned short* kh_b  = (unsigned short*)alloc((size_t)S_LEN * NKVH * DHEAD * 2);
    unsigned short* vT_b  = (unsigned short*)alloc((size_t)S_LEN * NKVH * DHEAD * 2);
    unsigned short* at_b  = (unsigned short*)alloc((size_t)S_LEN * NH   * DHEAD * 2);
    float*          x1_f  = (float*)alloc((size_t)S_LEN * DMODEL * 4);
    unsigned short* b_b   = (unsigned short*)alloc((size_t)S_LEN * DMODEL * 2);
    float*          g_f   = (float*)alloc((size_t)S_LEN * HID * 4);
    float*          u_f   = (float*)alloc((size_t)S_LEN * HID * 4);
    unsigned short* h_b   = (unsigned short*)alloc((size_t)S_LEN * HID * 2);

    // 1) a = rmsnorm(x) -> bf16
    rmsnorm_bf16_kernel<<<S_LEN, 256, 0, stream>>>(x, ln_w, a_b, DMODEL);

    // 2) q/k/v projections (fp32 out, bf16 WMMA inside)
    gemm_bf16_wmma_kernel<<<dim3(NH * DHEAD / BN, S_LEN / BM), 256, 0, stream>>>(
        a_b, wq, nullptr, q_f, S_LEN, NH * DHEAD, DMODEL);
    gemm_bf16_wmma_kernel<<<dim3(NKVH * DHEAD / BN, S_LEN / BM), 256, 0, stream>>>(
        a_b, wk, nullptr, k_f, S_LEN, NKVH * DHEAD, DMODEL);
    gemm_bf16_wmma_kernel<<<dim3(NKVH * DHEAD / BN, S_LEN / BM), 256, 0, stream>>>(
        a_b, wv, nullptr, v_f, S_LEN, NKVH * DHEAD, DMODEL);

    // 3) RoPE + per-head layouts
    {
        const int nq = S_LEN * NH * (DHEAD / 2);
        rope_transpose_kernel<<<(nq + 255) / 256, 256, 0, stream>>>(q_f, qh_b, pos, NH);
        const int nk = S_LEN * NKVH * (DHEAD / 2);
        rope_transpose_kernel<<<(nk + 255) / 256, 256, 0, stream>>>(k_f, kh_b, pos, NKVH);
        const int nv = S_LEN * NKVH * DHEAD;
        v_transpose_kernel<<<(nv + 255) / 256, 256, 0, stream>>>(v_f, vT_b);
    }

    // 4) causal flash attention -> bf16 [s][h*dh]
    flash_attn_kernel<<<dim3(S_LEN / 16, NH), 32, 0, stream>>>(qh_b, kh_b, vT_b, at_b);

    // 5) x1 = x + attn @ wo
    gemm_bf16_wmma_kernel<<<dim3(DMODEL / BN, S_LEN / BM), 256, 0, stream>>>(
        at_b, wo, x, x1_f, S_LEN, DMODEL, NH * DHEAD);

    // 6) b = rmsnorm(x1) -> bf16
    rmsnorm_bf16_kernel<<<S_LEN, 256, 0, stream>>>(x1_f, ffln_w, b_b, DMODEL);

    // 7) FFN: g = b@wg, u = b@w1, h = silu(g)*u
    gemm_bf16_wmma_kernel<<<dim3(HID / BN, S_LEN / BM), 256, 0, stream>>>(
        b_b, wg, nullptr, g_f, S_LEN, HID, DMODEL);
    gemm_bf16_wmma_kernel<<<dim3(HID / BN, S_LEN / BM), 256, 0, stream>>>(
        b_b, w1, nullptr, u_f, S_LEN, HID, DMODEL);
    {
        const long long np = (long long)S_LEN * HID / 2;
        silu_mul_kernel<<<(unsigned)((np + 255) / 256), 256, 0, stream>>>(
            g_f, u_f, (unsigned int*)h_b, np);
    }

    // 8) out = x1 + h @ w2
    gemm_bf16_wmma_kernel<<<dim3(DMODEL / BN, S_LEN / BM), 256, 0, stream>>>(
        h_b, w2, x1_f, out, S_LEN, DMODEL, HID);
}
